// GaussianFeatureLeverage_46686294507956
// MI455X (gfx1250) — matrix operand order repaired and checked by reading, not compiled, not run on hardware
//
#include <hip/hip_runtime.h>
#include <hip/hip_bf16.h>

typedef __attribute__((ext_vector_type(16))) _Float16 v16h;
typedef __attribute__((ext_vector_type(8)))  _Float16 v8h;
typedef __attribute__((ext_vector_type(8)))  float    v8f;
typedef __attribute__((ext_vector_type(4)))  unsigned int v4u;
typedef __attribute__((ext_vector_type(4)))  int      v4i;
typedef __attribute__((ext_vector_type(8)))  int      v8i;

#define H0   96
#define W0   320
#define HW0  (H0 * W0)
#define CPAD 288   // 263 channels padded to 9*32

// per-(mpair,kstep) weight slab: 2 subtiles x 9 taps x (16x32) f16 = 18432 B
#define SLAB_HALVES (2 * 9 * 512)
#define SLAB_BYTES  (SLAB_HALVES * 2)
#define SLAB_ELEM8  (SLAB_BYTES / 8)   // 2304 8-byte elements

#if defined(__has_builtin)
#if __has_builtin(__builtin_amdgcn_tensor_load_to_lds) && \
    __has_builtin(__builtin_amdgcn_s_wait_tensorcnt)
#define USE_TDM 1
#endif
#endif

// ---------------------------------------------------------------------------
// Activation: 0 none, 1 gelu(tanh), 2 elu, 3 sigmoid
// ---------------------------------------------------------------------------
__device__ __forceinline__ float act_apply(float x, int act) {
  switch (act) {
    case 1: {
      float x3 = x * x * x;
      return 0.5f * x * (1.0f + tanhf(0.7978845608028654f * (x + 0.044715f * x3)));
    }
    case 2: return x > 0.0f ? x : expm1f(x);
    case 3: return 1.0f / (1.0f + expf(-x));
    default: return x;
  }
}

// ---------------------------------------------------------------------------
// Pack f32 OIHW conv weights into f16 A-fragment order, M-pair major:
//   idx = (((mp*ksteps + ks)*2 + h)*9 + tap)*512 + mm*32 + cc
//   m = mp*32 + h*16 + mm ; c = ks*32 + cc ; zero outside (Cout, Cin)
// ---------------------------------------------------------------------------
__global__ void pack_weights_kernel(const float* __restrict__ Wg, int Cin, int Cout,
                                    int ksteps, long total, _Float16* __restrict__ out)
{
  long idx = (long)blockIdx.x * blockDim.x + threadIdx.x;
  if (idx >= total) return;
  int  cc  = (int)(idx & 31);
  int  mm  = (int)((idx >> 5) & 15);
  int  tap = (int)((idx >> 9) % 9);
  long rem = idx / (512 * 9);
  int  h   = (int)(rem & 1);
  long rem2 = rem >> 1;
  int  ks  = (int)(rem2 % ksteps);
  int  mp  = (int)(rem2 / ksteps);
  int m = mp * 32 + h * 16 + mm, c = ks * 32 + cc;
  float v = (m < Cout && c < Cin) ? Wg[(size_t)(m * Cin + c) * 9 + tap] : 0.0f;
  out[idx] = (_Float16)v;
}

// ---------------------------------------------------------------------------
// Implicit-GEMM 3x3 conv (pad=1, clamped borders) via WMMA f16.
// Register-blocked 2 M-tiles per wave: one B fragment feeds two WMMAs, halving
// global B traffic. Per kstep an 18 KB weight slab (both subtiles, all 9 taps)
// is DMA'd to LDS by the Tensor Data Mover (fallback: vectorized LDS copy).
// Tap loop fully unrolled: 9x (2 global_load_b128 + 4 ds_load_b128 + 2 wmma).
// outMode: 0 f32 CL(stride), 1 f16 CL(stride), 2 f32 channel-first.
// ---------------------------------------------------------------------------
__global__ __launch_bounds__(128)
void conv3x3_wmma_kernel(const _Float16* __restrict__ X, int H, int W,
                         int CinPad,
                         const _Float16* __restrict__ Wpk,
                         const float* __restrict__ Bias,
                         int Cout, int outStride,
                         void* __restrict__ Yv,
                         int act, int outMode)
{
  __shared__ __align__(16) _Float16 wlds[SLAB_HALVES];   // 18 KB

  const int HW   = H * W;
  const int lane = threadIdx.x & 31;
  const int wave = threadIdx.x >> 5;
  const int mp   = blockIdx.y;                 // M pair -> 32 output channels
  const int p0   = (blockIdx.x * 4 + wave) * 16;

  // A fragment (16x32 f16): lane<16 row=lane, K 0..7 & 16..23; lane>=16 K 8..15 & 24..31
  const int arow = lane & 15;
  const int kh   = (lane >> 4) * 8;
  // B fragment (32x16 f16): lane<16 col=lane K 0..7 & 16..23; lane>=16 K 8..15 & 24..31
  const int bcol = lane & 15;
  const int bk   = (lane >> 4) * 8;

  const int pix  = p0 + bcol;
  const int pixc = pix < HW ? pix : HW - 1;    // clamped; OOB columns never stored
  const int py   = pixc / W;
  const int px   = pixc % W;

  // hoist per-tap B base pointers (clamped 3x3 neighborhood)
  const _Float16* xptap[9];
#pragma unroll
  for (int tap = 0; tap < 9; ++tap) {
    const int dy = tap / 3 - 1, dx = tap % 3 - 1;
    int sy = py + dy; sy = sy < 0 ? 0 : (sy >= H ? H - 1 : sy);
    int sx = px + dx; sx = sx < 0 ? 0 : (sx >= W ? W - 1 : sx);
    xptap[tap] = X + (size_t)(sy * W + sx) * CinPad + bk;
  }

  const int ksteps = CinPad >> 5;
  const _Float16* wslab = Wpk + (size_t)mp * ksteps * SLAB_HALVES;
  const unsigned ldsoff = (unsigned)(uintptr_t)(void*)wlds;  // LDS aperture: low 32b = offset
  __builtin_prefetch(wslab, 0, 2);

  v8f acc0 = {};
  v8f acc1 = {};

  for (int ks = 0; ks < ksteps; ++ks) {
    __syncthreads();
#ifdef USE_TDM
    if (wave == 0) {
      unsigned long long ga = (unsigned long long)(uintptr_t)(wslab + (size_t)ks * SLAB_HALVES);
      // D# group0: count=1, lds_addr, global_addr, type=2
      v4u g0;
      g0[0] = 1u;
      g0[1] = ldsoff;
      g0[2] = (unsigned)ga;
      g0[3] = (unsigned)((ga >> 32) & 0x01FFFFFFu) | (2u << 30);
      // D# group1: data_size=8B, tensor SLAB_ELEM8 x 1, tile SLAB_ELEM8 x 1
      v8i g1;
      g1[0] = 0x00030000;                                  // mask=0, data_size=3 (8B)
      g1[1] = (int)(((unsigned)SLAB_ELEM8 & 0xFFFFu) << 16);  // tensor_dim0 lo16
      g1[2] = (int)(1u << 16);                             // tensor_dim0 hi16 | tensor_dim1 lo16
      g1[3] = (int)(((unsigned)SLAB_ELEM8 & 0xFFFFu) << 16);  // tensor_dim1 hi | tile_dim0
      g1[4] = 1;                                           // tile_dim1=1, tile_dim2=0
      g1[5] = SLAB_ELEM8;                                  // tensor_dim0_stride lo32
      g1[6] = 0;
      g1[7] = 0;
      v4i g2 = {}; v4i g3 = {};
#if defined(__clang_major__) && (__clang_major__ >= 23)
      v8i g4 = {};
      __builtin_amdgcn_tensor_load_to_lds(g0, g1, g2, g3, g4, 0);
#else
      __builtin_amdgcn_tensor_load_to_lds(g0, g1, g2, g3, 0);
#endif
      __builtin_amdgcn_s_wait_tensorcnt(0);
    }
#else
    {
      const v8h* src = (const v8h*)(wslab + (size_t)ks * SLAB_HALVES);
      v8h* dst = (v8h*)wlds;
      for (int t = threadIdx.x; t < SLAB_HALVES / 8; t += 128) dst[t] = src[t];
    }
#endif
    __syncthreads();

    const int c0 = ks << 5;
#pragma unroll
    for (int tap = 0; tap < 9; ++tap) {
      // B fragment (shared by both M subtiles)
      const _Float16* xp = xptap[tap] + c0;
      v8h b0 = *(const v8h*)(xp);
      v8h b1 = *(const v8h*)(xp + 16);
      v16h b;
#pragma unroll
      for (int i = 0; i < 8; ++i) { b[i] = b0[i]; b[8 + i] = b1[i]; }

      // A fragment, subtile 0
      {
        const _Float16* wp = &wlds[tap * 512 + arow * 32];
        v8h lo = *(const v8h*)(wp + kh);
        v8h hi = *(const v8h*)(wp + 16 + kh);
        v16h a;
#pragma unroll
        for (int i = 0; i < 8; ++i) { a[i] = lo[i]; a[8 + i] = hi[i]; }
        acc0 = __builtin_amdgcn_wmma_f32_16x16x32_f16(false, a, false, b,
                                                      (short)0, acc0, false, false);
      }
      // A fragment, subtile 1
      {
        const _Float16* wp = &wlds[9 * 512 + tap * 512 + arow * 32];
        v8h lo = *(const v8h*)(wp + kh);
        v8h hi = *(const v8h*)(wp + 16 + kh);
        v16h a;
#pragma unroll
        for (int i = 0; i < 8; ++i) { a[i] = lo[i]; a[8 + i] = hi[i]; }
        acc1 = __builtin_amdgcn_wmma_f32_16x16x32_f16(false, a, false, b,
                                                      (short)0, acc1, false, false);
      }
    }
  }

  // D layout: element r -> M = r + 8*(lane>=16), N = lane&15
  const int col  = lane & 15;
  const int mb   = (lane >> 4) * 8;
  const int opix = p0 + col;
  if (opix < HW) {
#pragma unroll
    for (int hsub = 0; hsub < 2; ++hsub) {
      const v8f* accp = hsub ? &acc1 : &acc0;
#pragma unroll
      for (int r = 0; r < 8; ++r) {
        int m = mp * 32 + hsub * 16 + mb + r;
        float bv = (m < Cout) ? Bias[m] : 0.0f;
        float v = act_apply((*accp)[r] + bv, act);
        if (m >= Cout) v = 0.0f;
        if (outMode == 1) {
          if (m < outStride) ((_Float16*)Yv)[(size_t)opix * outStride + m] = (_Float16)v;
        } else if (outMode == 0) {
          if (m < outStride) ((float*)Yv)[(size_t)opix * outStride + m] = v;
        } else {
          if (m < Cout) ((float*)Yv)[(size_t)m * HW + opix] = v;
        }
      }
    }
  }
}

// ---------------------------------------------------------------------------
// Build the 263-ch concatenated feature map (bilinear upsample) + depth.
// ---------------------------------------------------------------------------
__global__ void build_feature_kernel(const float* __restrict__ f0, const float* __restrict__ f1,
                                     const float* __restrict__ f2, const float* __restrict__ f3,
                                     const float* __restrict__ d0, const float* __restrict__ d1,
                                     const float* __restrict__ d2, const float* __restrict__ d3,
                                     const float* __restrict__ c0,
                                     _Float16* __restrict__ feat, float* __restrict__ depth)
{
  int pix = blockIdx.x * blockDim.x + threadIdx.x;
  if (pix >= HW0) return;
  int py = pix / W0, px = pix % W0;
  _Float16* dst = feat + (size_t)pix * CPAD;

  const float* srcs[4] = { f0, f1, f2, f3 };
  const float* dsp[4]  = { d0, d1, d2, d3 };
  const int    chn[4]  = { 32, 32, 64, 128 };

  int cbase = 0;
  float dvals[4];
#pragma unroll
  for (int lev = 0; lev < 4; ++lev) {
    int hs = H0 >> (lev + 2), ws = W0 >> (lev + 2);
    float syf = ((float)py + 0.5f) * (float)hs / (float)H0 - 0.5f;
    float sxf = ((float)px + 0.5f) * (float)ws / (float)W0 - 0.5f;
    int y0 = (int)floorf(syf), x0 = (int)floorf(sxf);
    float wy = syf - (float)y0, wx = sxf - (float)x0;
    int y1 = y0 + 1, x1 = x0 + 1;
    y0 = y0 < 0 ? 0 : (y0 >= hs ? hs - 1 : y0);
    y1 = y1 < 0 ? 0 : (y1 >= hs ? hs - 1 : y1);
    x0 = x0 < 0 ? 0 : (x0 >= ws ? ws - 1 : x0);
    x1 = x1 < 0 ? 0 : (x1 >= ws ? ws - 1 : x1);
    int i00 = y0 * ws + x0, i01 = y0 * ws + x1, i10 = y1 * ws + x0, i11 = y1 * ws + x1;
    float w00 = (1.f - wy) * (1.f - wx), w01 = (1.f - wy) * wx;
    float w10 = wy * (1.f - wx), w11 = wy * wx;
    int plane = hs * ws;
    const float* s = srcs[lev];
    for (int c = 0; c < chn[lev]; ++c) {
      const float* p = s + (size_t)c * plane;
      dst[cbase + c] = (_Float16)(p[i00] * w00 + p[i01] * w01 + p[i10] * w10 + p[i11] * w11);
    }
    cbase += chn[lev];
    const float* p = dsp[lev];
    dvals[lev] = p[i00] * w00 + p[i01] * w01 + p[i10] * w10 + p[i11] * w11;
  }
#pragma unroll
  for (int lev = 0; lev < 4; ++lev) dst[256 + lev] = (_Float16)dvals[lev];
  for (int c = 0; c < 3; ++c) dst[260 + c] = (_Float16)c0[(size_t)c * HW0 + pix];
  for (int c = 263; c < CPAD; ++c) dst[c] = (_Float16)0.0f;

  float scaled = 0.01f + (10.0f - 0.01f) * dvals[0];
  float dpt = 1.0f / scaled;
  dpt = fminf(fmaxf(dpt, 0.1f), 100.0f);
  depth[pix] = dpt;
}

// ---------------------------------------------------------------------------
// Per-gaussian projection / 2D covariance / alpha.
// ---------------------------------------------------------------------------
__device__ bool splat_compute(int p, const float* __restrict__ rot,
                              const float* __restrict__ scl, const float* __restrict__ opa,
                              const float* __restrict__ depth, const float* __restrict__ K,
                              int ki, int h, int w, int* pid_out, float* alpha_out)
{
  float z = depth[p];
  int pyf = p / W0, pxf = p % W0;
  float fx0 = K[0], cx0 = K[2], fy0 = K[4], cy0 = K[5];
  float x = z * ((float)pxf - cx0) / fx0;
  float y = z * ((float)pyf - cy0) / fy0;
  const float* Ki = K + (size_t)ki * 9;
  float fx = Ki[0], cx = Ki[2], fy = Ki[4], cy = Ki[5];
  float zc = fmaxf(z, 1e-6f);
  float u = fx * x / zc + cx, v = fy * y / zc + cy;

  float q0 = rot[(size_t)p * 4 + 0], q1 = rot[(size_t)p * 4 + 1];
  float q2 = rot[(size_t)p * 4 + 2], q3 = rot[(size_t)p * 4 + 3];
  float qn = fmaxf(sqrtf(q0 * q0 + q1 * q1 + q2 * q2 + q3 * q3), 1e-12f);
  q0 /= qn; q1 /= qn; q2 /= qn; q3 /= qn;
  float s0 = fabsf(scl[(size_t)p * 4 + 0]);
  float s1 = fabsf(scl[(size_t)p * 4 + 1]);
  float s2 = fabsf(scl[(size_t)p * 4 + 2]);
  float ss0 = s0 * s0, ss1 = s1 * s1, ss2 = s2 * s2;

  float R[3][3] = {
    { 1.f - 2.f * (q2 * q2 + q3 * q3), 2.f * (q1 * q2 - q0 * q3), 2.f * (q1 * q3 + q0 * q2) },
    { 2.f * (q1 * q2 + q0 * q3), 1.f - 2.f * (q1 * q1 + q3 * q3), 2.f * (q2 * q3 - q0 * q1) },
    { 2.f * (q1 * q3 - q0 * q2), 2.f * (q2 * q3 + q0 * q1), 1.f - 2.f * (q1 * q1 + q2 * q2) } };
  float cov3[3][3];
#pragma unroll
  for (int i = 0; i < 3; ++i)
#pragma unroll
    for (int j = 0; j < 3; ++j)
      cov3[i][j] = R[i][0] * ss0 * R[j][0] + R[i][1] * ss1 * R[j][1] + R[i][2] * ss2 * R[j][2];

  float j00 = fx / zc, j02 = -fx * x / (zc * zc);
  float j11 = fy / zc, j12 = -fy * y / (zc * zc);
  float t00 = j00 * cov3[0][0] + j02 * cov3[2][0];
  float t01 = j00 * cov3[0][1] + j02 * cov3[2][1];
  float t02 = j00 * cov3[0][2] + j02 * cov3[2][2];
  float t10 = j11 * cov3[1][0] + j12 * cov3[2][0];
  float t11 = j11 * cov3[1][1] + j12 * cov3[2][1];
  float t12 = j11 * cov3[1][2] + j12 * cov3[2][2];
  float c00 = t00 * j00 + t02 * j02;
  float c01 = t01 * j11 + t02 * j12;
  float c11 = t11 * j11 + t12 * j12;

  float a  = c00 + 0.3f, cc = c11 + 0.3f, b = c01;
  float det = fmaxf(a * cc - b * b, 1e-8f);
  int ppx = (int)floorf(u), ppy = (int)floorf(v);
  float dx = u - ((float)ppx + 0.5f), dy = v - ((float)ppy + 0.5f);
  float power = -0.5f * (cc * dx * dx - 2.f * b * dx * dy + a * dy * dy) / det;
  float alpha = fminf(0.99f, opa[p] * expf(fminf(power, 0.0f)));
  bool valid = (z > 0.1f) && (z < 100.f) && ppx >= 0 && ppx < w && ppy >= 0 && ppy < h &&
               (alpha > 1.0f / 255.0f);
  *pid_out = ppy * w + ppx;
  *alpha_out = alpha;
  return valid;
}

__global__ void raster_pass1_kernel(const float* __restrict__ rot, const float* __restrict__ scl,
                                    const float* __restrict__ opa, const float* __restrict__ depth,
                                    const float* __restrict__ K, int ki, int h, int w,
                                    float* __restrict__ Sbuf)
{
  int p = blockIdx.x * blockDim.x + threadIdx.x;
  if (p >= HW0) return;
  int pid; float alpha;
  if (splat_compute(p, rot, scl, opa, depth, K, ki, h, w, &pid, &alpha))
    atomicAdd(&Sbuf[pid], log1pf(-alpha));
}

__global__ void raster_pass2_kernel(const float* __restrict__ rot, const float* __restrict__ scl,
                                    const float* __restrict__ opa, const float* __restrict__ depth,
                                    const float* __restrict__ K, int ki, int h, int w,
                                    const float* __restrict__ gsf, const float* __restrict__ Sbuf,
                                    float* __restrict__ ras)
{
  int p = blockIdx.x * blockDim.x + threadIdx.x;
  if (p >= HW0) return;
  int pid; float alpha;
  if (splat_compute(p, rot, scl, opa, depth, K, ki, h, w, &pid, &alpha)) {
    float l = log1pf(-alpha);
    float T = expf(Sbuf[pid] - l);     // product of (1-alpha) of co-resident splats
    float wgt = alpha * T;
    const float* g = gsf + (size_t)p * 64;
    float* o = ras + (size_t)pid * 64;
    for (int c = 0; c < 64; ++c) atomicAdd(&o[c], wgt * g[c]);
  }
}

// ---------------------------------------------------------------------------
// Pack [ras(64) | feat_i(chi) | color_i(3) | disp_i(1) | zeros] as f16 CL.
// ---------------------------------------------------------------------------
__global__ void pack_fused_kernel(const float* __restrict__ ras, const float* __restrict__ feat_i,
                                  const float* __restrict__ col_i, const float* __restrict__ disp_i,
                                  int hw, int chi, int cpad, _Float16* __restrict__ out)
{
  int pix = blockIdx.x * blockDim.x + threadIdx.x;
  if (pix >= hw) return;
  _Float16* d = out + (size_t)pix * cpad;
  const float* r = ras + (size_t)pix * 64;
  for (int c = 0; c < 64; ++c) d[c] = (_Float16)r[c];
  for (int c = 0; c < chi; ++c) d[64 + c] = (_Float16)feat_i[(size_t)c * hw + pix];
  for (int c = 0; c < 3; ++c) d[64 + chi + c] = (_Float16)col_i[(size_t)c * hw + pix];
  d[64 + chi + 3] = (_Float16)disp_i[pix];
  for (int c = 64 + chi + 4; c < cpad; ++c) d[c] = (_Float16)0.0f;
}

// ---------------------------------------------------------------------------
// Host launcher
// ---------------------------------------------------------------------------
static void launch_conv(const _Float16* X, int H, int W, int CinPad,
                        const _Float16* Wpk, const float* Bs, int Cout, int outStride,
                        void* Y, int act, int outMode, int mpairs, hipStream_t s)
{
  int HW = H * W;
  int ptiles = (HW + 15) / 16;
  int gx = (ptiles + 3) / 4;
  dim3 grid(gx, mpairs), block(128);
  hipLaunchKernelGGL(conv3x3_wmma_kernel, grid, block, 0, s,
                     X, H, W, CinPad, Wpk, Bs, Cout, outStride, Y, act, outMode);
}

extern "C" void kernel_launch(void* const* d_in, const int* in_sizes, int n_in,
                              void* d_out, int out_size, void* d_ws, size_t ws_size,
                              hipStream_t stream)
{
  const float* feat_in[4] = { (const float*)d_in[0], (const float*)d_in[3],
                              (const float*)d_in[6], (const float*)d_in[9] };
  const float* disp_in[4] = { (const float*)d_in[1], (const float*)d_in[4],
                              (const float*)d_in[7], (const float*)d_in[10] };
  const float* col_in[4]  = { (const float*)d_in[2], (const float*)d_in[5],
                              (const float*)d_in[8], (const float*)d_in[11] };
  const float* color0 = (const float*)d_in[12];
  const float* Kmat   = (const float*)d_in[13];
  const float* rot_w1 = (const float*)d_in[14]; const float* rot_b1 = (const float*)d_in[15];
  const float* rot_w2 = (const float*)d_in[16]; const float* rot_b2 = (const float*)d_in[17];
  const float* scl_w1 = (const float*)d_in[18]; const float* scl_b1 = (const float*)d_in[19];
  const float* scl_w2 = (const float*)d_in[20]; const float* scl_b2 = (const float*)d_in[21];
  const float* opa_w1 = (const float*)d_in[22]; const float* opa_b1 = (const float*)d_in[23];
  const float* opa_w2 = (const float*)d_in[24]; const float* opa_b2 = (const float*)d_in[25];
  const float* lev_w1 = (const float*)d_in[26]; const float* lev_b1 = (const float*)d_in[27];
  const float* lev_w2 = (const float*)d_in[28]; const float* lev_b2 = (const float*)d_in[29];
  const float* res_w[4] = { (const float*)d_in[30], (const float*)d_in[32],
                            (const float*)d_in[34], (const float*)d_in[36] };
  const float* res_b[4] = { (const float*)d_in[31], (const float*)d_in[33],
                            (const float*)d_in[35], (const float*)d_in[37] };

  // workspace carve-up
  char* ws = (char*)d_ws;
  size_t off = 0;
  auto alloc = [&](size_t bytes) -> void* {
    void* p = ws + off;
    off = (off + bytes + 255) & ~(size_t)255;
    return p;
  };
  _Float16* feat16  = (_Float16*)alloc((size_t)HW0 * CPAD * 2);
  _Float16* lev16   = (_Float16*)alloc((size_t)HW0 * CPAD * 2);
  _Float16* hid16   = (_Float16*)alloc((size_t)HW0 * 32 * 2);
  float*    rot_out = (float*)alloc((size_t)HW0 * 4 * 4);
  float*    scl_out = (float*)alloc((size_t)HW0 * 4 * 4);
  float*    opa_out = (float*)alloc((size_t)HW0 * 1 * 4);
  float*    gsf     = (float*)alloc((size_t)HW0 * 64 * 4);
  float*    depth   = (float*)alloc((size_t)HW0 * 4);
  float*    Sbuf    = (float*)alloc((size_t)1920 * 4);
  float*    ras     = (float*)alloc((size_t)1920 * 64 * 4);
  _Float16* fused16 = (_Float16*)alloc((size_t)1920 * 224 * 2);

  // weight packing (f32 OIHW -> f16 A-fragment M-pair order), once per launch
  auto pack = [&](const float* Wg, int Cin, int CinPad, int Cout, int mpairs) -> _Float16* {
    int ksteps = CinPad / 32;
    long total = (long)mpairs * ksteps * SLAB_HALVES;
    _Float16* dst = (_Float16*)alloc((size_t)total * 2);
    hipLaunchKernelGGL(pack_weights_kernel, dim3((unsigned)((total + 255) / 256)), dim3(256),
                       0, stream, Wg, Cin, Cout, ksteps, total, dst);
    return dst;
  };

  const int CH_IN[4]  = { 32, 32, 64, 128 };
  const int CH_OUT[4] = { 18, 36, 72, 144 };
  const int CPADI[4]  = { 128, 128, 160, 224 };

  _Float16* pk_rot1 = pack(rot_w1, 263, CPAD, 8, 1);
  _Float16* pk_rot2 = pack(rot_w2, 8, 32, 4, 1);
  _Float16* pk_scl1 = pack(scl_w1, 263, CPAD, 6, 1);
  _Float16* pk_scl2 = pack(scl_w2, 6, 32, 3, 1);
  _Float16* pk_opa1 = pack(opa_w1, 263, CPAD, 2, 1);
  _Float16* pk_opa2 = pack(opa_w2, 2, 32, 1, 1);
  _Float16* pk_lev1 = pack(lev_w1, 263, CPAD, 263, 9);   // covers stride 288 = 9 pairs
  _Float16* pk_lev2 = pack(lev_w2, 263, CPAD, 64, 2);
  _Float16* pk_res[4];
  for (int i = 0; i < 4; ++i) {
    int cin_i = 64 + CH_IN[i] + 3 + 1;
    pk_res[i] = pack(res_w[i], cin_i, CPADI[i], CH_OUT[i], (CH_OUT[i] + 31) / 32);
  }

  // 1) feature concat + depth
  hipLaunchKernelGGL(build_feature_kernel, dim3((HW0 + 255) / 256), dim3(256), 0, stream,
                     feat_in[0], feat_in[1], feat_in[2], feat_in[3],
                     disp_in[0], disp_in[1], disp_in[2], disp_in[3],
                     color0, feat16, depth);

  // 2) heads (hidden layers share hid16, f16 CL stride 32 = exactly 1 M-pair)
  launch_conv(feat16, H0, W0, CPAD, pk_rot1, rot_b1, 8, 32, hid16, 1, 1, 1, stream);
  launch_conv(hid16,  H0, W0,   32, pk_rot2, rot_b2, 4,  4, rot_out, 0, 0, 1, stream);
  launch_conv(feat16, H0, W0, CPAD, pk_scl1, scl_b1, 6, 32, hid16, 1, 1, 1, stream);
  launch_conv(hid16,  H0, W0,   32, pk_scl2, scl_b2, 3,  4, scl_out, 0, 0, 1, stream);
  launch_conv(feat16, H0, W0, CPAD, pk_opa1, opa_b1, 2, 32, hid16, 1, 1, 1, stream);
  launch_conv(hid16,  H0, W0,   32, pk_opa2, opa_b2, 1,  1, opa_out, 3, 0, 1, stream);

  // 3) level feature convs (the big ones)
  launch_conv(feat16, H0, W0, CPAD, pk_lev1, lev_b1, 263, CPAD, lev16, 2, 1, 9, stream);
  launch_conv(lev16,  H0, W0, CPAD, pk_lev2, lev_b2, 64, 64, gsf, 2, 0, 2, stream);

  // 4) per-scale rasterize + fuse + output conv
  const size_t out_off[4] = { 0, 34560, 51840, 60480 };
  float* outp = (float*)d_out;

  for (int i = 0; i < 4; ++i) {
    int h = H0 >> (i + 2), w = W0 >> (i + 2);
    int hw = h * w;

    hipMemsetAsync(Sbuf, 0, (size_t)hw * 4, stream);
    hipMemsetAsync(ras, 0, (size_t)hw * 64 * 4, stream);

    hipLaunchKernelGGL(raster_pass1_kernel, dim3((HW0 + 255) / 256), dim3(256), 0, stream,
                       rot_out, scl_out, opa_out, depth, Kmat, i + 2, h, w, Sbuf);
    hipLaunchKernelGGL(raster_pass2_kernel, dim3((HW0 + 255) / 256), dim3(256), 0, stream,
                       rot_out, scl_out, opa_out, depth, Kmat, i + 2, h, w, gsf, Sbuf, ras);

    hipLaunchKernelGGL(pack_fused_kernel, dim3((hw + 127) / 128), dim3(128), 0, stream,
                       ras, feat_in[i], col_in[i], disp_in[i], hw, CH_IN[i], CPADI[i], fused16);

    launch_conv(fused16, h, w, CPADI[i], pk_res[i], res_b[i],
                CH_OUT[i], 0, outp + out_off[i], 2, 2, (CH_OUT[i] + 31) / 32, stream);
  }
}
